// EncoderLayer_17575006175315
// MI455X (gfx1250) — matrix operand order repaired and checked by reading, not compiled, not run on hardware
//
#include <hip/hip_runtime.h>
#include <hip/hip_bf16.h>

typedef __bf16 bf16_t;
typedef __bf16 bf16x8  __attribute__((ext_vector_type(8)));
typedef __bf16 bf16x16 __attribute__((ext_vector_type(16)));
typedef float  floatx8 __attribute__((ext_vector_type(8)));

__device__ __forceinline__ floatx8 wmma_bf16f32(bf16x16 a, bf16x16 b, floatx8 c) {
  // v_wmma_f32_16x16x32_bf16: (neg_a, A, neg_b, B, c_mod, C, reuse_a, reuse_b)
  return __builtin_amdgcn_wmma_f32_16x16x32_bf16(false, a, false, b, (short)0, c,
                                                 false, false);
}

// A-operand fragment (16x32 bf16, ISA 7.12.2): lane<16 -> K 0..7 & 16..23,
// lane>=16 -> K 8..15 & 24..31; row = lane%16 (rowPtr pre-selected).
__device__ __forceinline__ bf16x16 frag_a(const bf16_t* rowPtr, int lane) {
  const int o0 = (lane < 16) ? 0 : 8;
  bf16x8 lo = *(const bf16x8*)(rowPtr + o0);
  bf16x8 hi = *(const bf16x8*)(rowPtr + o0 + 16);
  return __builtin_shufflevector(lo, hi, 0,1,2,3,4,5,6,7,8,9,10,11,12,13,14,15);
}

// B-operand fragment from B^T ([N][K]) storage: lane<16 -> K 0..7 & 16..23,
// lane>=16 -> K 8..15 & 24..31; col = lane%16 (rowPtr = row n base + kBase).
__device__ __forceinline__ bf16x16 frag_b(const bf16_t* rowPtr, int lane) {
  const int o0 = (lane < 16) ? 0 : 8;
  bf16x8 lo = *(const bf16x8*)(rowPtr + o0);
  bf16x8 hi = *(const bf16x8*)(rowPtr + o0 + 16);
  return __builtin_shufflevector(lo, hi, 0,1,2,3,4,5,6,7,8,9,10,11,12,13,14,15);
}

// ---- CDNA5 async global->LDS copy (ASYNCcnt-tracked DMA, bypasses VGPRs) ----
__device__ __forceinline__ void async_copy_b128(const bf16_t* g, bf16_t* l) {
  const unsigned ldsOff =
      (unsigned)(uintptr_t)(__attribute__((address_space(3))) bf16_t*)l;
  asm volatile("global_load_async_to_lds_b128 %0, %1, off"
               :
               : "v"(ldsOff), "v"((unsigned long long)(uintptr_t)g)
               : "memory");
}

__device__ __forceinline__ void wait_async0() {
  asm volatile("s_wait_asynccnt 0x0" ::: "memory");
}

// ---------------- generic batched WMMA GEMM: C = alpha*A*B^T (+bias) ----------------
// A: [M,K] bf16 (async double-buffered LDS staging) or f32 via Af (sync staging with
// on-the-fly bf16 conversion). Bm: [N,K] bf16.
// Per-batch offsets: off = (z/innerCnt)*sXo + (z%innerCnt)*sXi.
// Block: 256 thr = 8 waves as WM x WN grid of 32x32 wave tiles.
template<int WM, int WN>
__global__ __launch_bounds__(256) void gemm_wmma(
    const bf16_t* __restrict__ A, const float* __restrict__ Af,
    long lda, long sAo, long sAi,
    const bf16_t* __restrict__ Bm, long ldb, long sBo, long sBi,
    float* __restrict__ Cf, bf16_t* __restrict__ Cb,
    long ldc, long sCo, long sCi,
    const float* __restrict__ bias, float alpha, int relu,
    int transStore, long ldt, int K, int innerCnt)
{
  constexpr int BM = WM * 32, BN = WN * 32;
  __shared__ __align__(64) bf16_t As[2][BM][32];
  __shared__ __align__(64) bf16_t Bs[2][BN][32];
  const int tid  = threadIdx.x;
  const int wid  = tid >> 5, lane = tid & 31;
  const int wm   = wid / WN, wn = wid % WN;
  const int z    = blockIdx.z;
  const int zo   = z / innerCnt, zi = z % innerCnt;
  const long offA = (long)zo * sAo + (long)zi * sAi;
  const long offB = (long)zo * sBo + (long)zi * sBi;
  const long offC = (long)zo * sCo + (long)zi * sCi;
  const int m0 = blockIdx.y * BM;
  const int n0 = blockIdx.x * BN;
  const int KT = K >> 5;

  // issue this thread's share of async copies for K-step kt into buffer kt&1
  auto stage_async = [&](int kt) {
    const int buf = kt & 1;
    const int k0 = kt << 5;
    for (int idx = tid; idx < BM * 4; idx += 256) {
      const int r = idx >> 2, c = (idx & 3) * 8;
      async_copy_b128(A + offA + (long)(m0 + r) * lda + (k0 + c), &As[buf][r][c]);
    }
    for (int idx = tid; idx < BN * 4; idx += 256) {
      const int r = idx >> 2, c = (idx & 3) * 8;
      async_copy_b128(Bm + offB + (long)(n0 + r) * ldb + (k0 + c), &Bs[buf][r][c]);
    }
  };

  floatx8 acc[2][2] = {};
  if (!Af) stage_async(0);           // uniform branch: prime the pipeline

  for (int kt = 0; kt < KT; ++kt) {
    int buf;
    if (!Af) {
      // async double-buffered path: wait own DMAs, barrier for others',
      // then immediately launch next panel while computing this one.
      wait_async0();
      __syncthreads();
      if (kt + 1 < KT) stage_async(kt + 1);
      buf = kt & 1;
    } else {
      // sync path with f32 -> bf16 conversion on the LDS fill (attn operand)
      if (kt > 0) __syncthreads();
      const int k0 = kt << 5;
      for (int idx = tid; idx < BM * 4; idx += 256) {
        const int r = idx >> 2, c = (idx & 3) * 8;
        const float* p = Af + offA + (long)(m0 + r) * lda + (k0 + c);
        bf16x8 v;
        #pragma unroll
        for (int j = 0; j < 8; ++j) v[j] = (bf16_t)p[j];
        *(bf16x8*)(&As[0][r][c]) = v;
      }
      for (int idx = tid; idx < BN * 4; idx += 256) {
        const int r = idx >> 2, c = (idx & 3) * 8;
        const long g = offB + (long)(n0 + r) * ldb + (k0 + c);
        *(bf16x8*)(&Bs[0][r][c]) = *(const bf16x8*)(Bm + g);
      }
      __syncthreads();
      buf = 0;
    }

    const bf16x16 a0 = frag_a(&As[buf][wm * 32 + (lane & 15)][0], lane);
    const bf16x16 a1 = frag_a(&As[buf][wm * 32 + 16 + (lane & 15)][0], lane);
    const bf16x16 b0 = frag_b(&Bs[buf][wn * 32 + (lane & 15)][0], lane);
    const bf16x16 b1 = frag_b(&Bs[buf][wn * 32 + 16 + (lane & 15)][0], lane);
    acc[0][0] = wmma_bf16f32(a0, b0, acc[0][0]);
    acc[0][1] = wmma_bf16f32(a0, b1, acc[0][1]);
    acc[1][0] = wmma_bf16f32(a1, b0, acc[1][0]);
    acc[1][1] = wmma_bf16f32(a1, b1, acc[1][1]);
  }

  // epilogue: C layout -> element i: row = tile_m + i + (lane<16?0:8), col = lane%16
  #pragma unroll
  for (int tm = 0; tm < 2; ++tm) {
    #pragma unroll
    for (int tn = 0; tn < 2; ++tn) {
      const int col = n0 + wn * 32 + tn * 16 + (lane & 15);
      const float bv = bias ? bias[col] : 0.0f;
      #pragma unroll
      for (int i = 0; i < 8; ++i) {
        const int row = m0 + wm * 32 + tm * 16 + i + ((lane < 16) ? 0 : 8);
        float v = acc[tm][tn][i] * alpha + bv;
        if (relu) v = fmaxf(v, 0.0f);
        if (Cf) Cf[offC + (long)row * ldc + col] = v;
        if (Cb) {
          if (transStore) Cb[offC + (long)col * ldt + row] = (bf16_t)v;
          else            Cb[offC + (long)row * ldc + col] = (bf16_t)v;
        }
      }
    }
  }
}

// ---------------- fused per-head FFN: ff = W2 * relu(W1*x + b1) + b2 ----------------
// Block: 16 tokens x 1 head; hidden kept in LDS in 1024-wide chunks.
__global__ __launch_bounds__(256) void ff_fused(
    const bf16_t* __restrict__ x1b, const bf16_t* __restrict__ w1b,
    const float* __restrict__ b1, const bf16_t* __restrict__ w2b,
    const float* __restrict__ b2, float* __restrict__ ffout)
{
  __shared__ __align__(64) bf16_t xs[16][64];
  __shared__ __align__(64) bf16_t hs[16][1024];
  __shared__ __align__(16) float  red[2][16][64];
  const int tb  = blockIdx.x * 16;
  const int h   = blockIdx.y;
  const int tid = threadIdx.x, wid = tid >> 5, lane = tid & 31;

  for (int idx = tid; idx < 16 * 8; idx += 256) {
    const int r = idx >> 3, c = (idx & 7) * 8;
    *(bf16x8*)(&xs[r][c]) =
        *(const bf16x8*)(x1b + (long)(tb + r) * 1024 + h * 64 + c);
  }
  __syncthreads();
  const bf16x16 a0 = frag_a(&xs[lane & 15][0], lane);   // K 0..31
  const bf16x16 a1 = frag_a(&xs[lane & 15][32], lane);  // K 32..63
  const bf16_t* W1h = w1b + (long)h * 4096 * 64;        // [FF, hd] = [N,K]
  const bf16_t* W2h = w2b + (long)h * 64 * 4096;        // [hd, FF] = [N,K]
  const int ntile = wid & 3, kh = wid >> 2;

  floatx8 acc2 = {};
  for (int chunk = 0; chunk < 4; ++chunk) {
    const int nc0 = chunk * 1024;
    // phase 1: hidden chunk, 8 waves x 8 n-tiles
    for (int t = 0; t < 8; ++t) {
      const int nl = wid * 128 + t * 16;
      const bf16_t* bp = W1h + (long)(nc0 + nl + (lane & 15)) * 64;
      __builtin_prefetch(bp + 16 * 64, 0, 0);   // global_prefetch next n-tile row
      floatx8 acc = {};
      acc = wmma_bf16f32(a0, frag_b(bp, lane), acc);
      acc = wmma_bf16f32(a1, frag_b(bp + 32, lane), acc);
      const int cl = nl + (lane & 15);
      const float bb = b1[(long)h * 4096 + nc0 + cl];
      #pragma unroll
      for (int i = 0; i < 8; ++i) {
        const int r = i + ((lane < 16) ? 0 : 8);
        hs[r][cl] = (bf16_t)fmaxf(acc[i] + bb, 0.0f);
      }
    }
    __syncthreads();
    // phase 2: accumulate this K-chunk into ff (4 n-tiles x 2 K-halves over waves)
    const bf16_t* brow =
        W2h + (long)(ntile * 16 + (lane & 15)) * 4096 + nc0 + kh * 512;
    for (int k = 0; k < 512; k += 32) {
      __builtin_prefetch(brow + k + 64, 0, 0);  // pull next W2 line toward WGP
      const bf16x16 af = frag_a(&hs[lane & 15][kh * 512 + k], lane);
      acc2 = wmma_bf16f32(af, frag_b(brow + k, lane), acc2);
    }
    __syncthreads();
  }
  {
    const int col = ntile * 16 + (lane & 15);
    #pragma unroll
    for (int i = 0; i < 8; ++i)
      red[kh][i + ((lane < 16) ? 0 : 8)][col] = acc2[i];
  }
  __syncthreads();
  for (int idx = tid; idx < 16 * 64; idx += 256) {
    const int r = idx >> 6, c = idx & 63;
    ffout[(long)(tb + r) * 1024 + h * 64 + c] =
        red[0][r][c] + red[1][r][c] + b2[(long)h * 64 + c];
  }
}

// ---------------- softmax over rows of length S, in place (already scaled) ----------
__global__ __launch_bounds__(256) void softmax_rows(float* __restrict__ attn, int S)
{
  float* p = attn + (long)blockIdx.x * S;
  __shared__ float red[256];
  const int tid = threadIdx.x;
  float m = -3.402823466e+38f;
  for (int c = tid; c < S; c += 256) m = fmaxf(m, p[c]);
  red[tid] = m; __syncthreads();
  for (int o = 128; o > 0; o >>= 1) {
    if (tid < o) red[tid] = fmaxf(red[tid], red[tid + o]);
    __syncthreads();
  }
  m = red[0]; __syncthreads();
  float s = 0.0f;
  for (int c = tid; c < S; c += 256) { float e = __expf(p[c] - m); p[c] = e; s += e; }
  red[tid] = s; __syncthreads();
  for (int o = 128; o > 0; o >>= 1) {
    if (tid < o) red[tid] += red[tid + o];
    __syncthreads();
  }
  const float inv = 1.0f / red[0]; __syncthreads();
  for (int c = tid; c < S; c += 256) p[c] *= inv;
}

// ---------------- LayerNorm(a+b) * g + be -> outF (and optional bf16 copy) ----------
__global__ __launch_bounds__(256) void layernorm_rows(
    const float* __restrict__ a, const float* __restrict__ b,
    const float* __restrict__ g, const float* __restrict__ be,
    float* __restrict__ outF, bf16_t* __restrict__ outB, int D)
{
  const long row = blockIdx.x;
  const float* pa = a + row * D;
  const float* pb = b + row * D;
  __shared__ float red[256];
  const int tid = threadIdx.x;
  float s = 0.0f;
  for (int c = tid; c < D; c += 256) s += pa[c] + pb[c];
  red[tid] = s; __syncthreads();
  for (int o = 128; o > 0; o >>= 1) { if (tid < o) red[tid] += red[tid + o]; __syncthreads(); }
  const float mean = red[0] / (float)D; __syncthreads();
  float v = 0.0f;
  for (int c = tid; c < D; c += 256) { float d = pa[c] + pb[c] - mean; v += d * d; }
  red[tid] = v; __syncthreads();
  for (int o = 128; o > 0; o >>= 1) { if (tid < o) red[tid] += red[tid + o]; __syncthreads(); }
  const float rstd = rsqrtf(red[0] / (float)D + 1e-5f); __syncthreads();
  for (int c = tid; c < D; c += 256) {
    const float val = (pa[c] + pb[c] - mean) * rstd * g[c] + be[c];
    if (outF) outF[row * D + c] = val;
    if (outB) outB[row * D + c] = (bf16_t)val;
  }
}

// ---------------- layout converters ----------------
__global__ __launch_bounds__(256) void cvt_f32_bf16(
    const float* __restrict__ in, bf16_t* __restrict__ out, long n)
{
  const long i = (long)blockIdx.x * 256 + threadIdx.x;
  if (i < n) out[i] = (bf16_t)in[i];
}

// in: [R,C] f32 row-major -> out: [C,R] bf16 row-major (weight -> [N,K])
__global__ __launch_bounds__(256) void transpose_f32_bf16(
    const float* __restrict__ in, bf16_t* __restrict__ out, int R, int C)
{
  const long i = (long)blockIdx.x * 256 + threadIdx.x;
  if (i < (long)R * C) {
    const int r = (int)(i / C), c = (int)(i % C);
    out[(long)c * R + r] = (bf16_t)in[i];
  }
}

extern "C" void kernel_launch(void* const* d_in, const int* in_sizes, int n_in,
                              void* d_out, int out_size, void* d_ws, size_t ws_size,
                              hipStream_t stream) {
  (void)in_sizes; (void)n_in; (void)out_size; (void)ws_size;
  const float* x   = (const float*)d_in[0];
  const float* Wq  = (const float*)d_in[1];
  const float* bq  = (const float*)d_in[2];
  const float* Wk  = (const float*)d_in[3];
  const float* bk  = (const float*)d_in[4];
  const float* Wv  = (const float*)d_in[5];
  const float* bv  = (const float*)d_in[6];
  const float* Wo  = (const float*)d_in[7];
  const float* bo  = (const float*)d_in[8];
  const float* W1  = (const float*)d_in[9];
  const float* b1  = (const float*)d_in[10];
  const float* W2  = (const float*)d_in[11];
  const float* b2  = (const float*)d_in[12];
  const float* g1  = (const float*)d_in[13];
  const float* be1 = (const float*)d_in[14];
  const float* g2  = (const float*)d_in[15];
  const float* be2 = (const float*)d_in[16];

  const int D = 1024, S = 2048, Bn = 2, H = 16, hd = 64, FFD = 4096;
  const long MT = (long)Bn * S;              // 4096 tokens
  float* yOut    = (float*)d_out;            // [MT, D]
  float* attnOut = yOut + MT * D;            // [Bn,H,S,S]

  char* ws = (char*)d_ws;
  size_t wo = 0;
  auto take = [&](size_t bytes) -> char* {
    char* p = ws + wo; wo = (wo + bytes + 255) & ~(size_t)255; return p;
  };
  bf16_t* xb    = (bf16_t*)take((size_t)MT * D * 2);
  bf16_t* wqT   = (bf16_t*)take((size_t)D * D * 2);
  bf16_t* wkT   = (bf16_t*)take((size_t)D * D * 2);
  bf16_t* wvT   = (bf16_t*)take((size_t)D * D * 2);
  bf16_t* woT   = (bf16_t*)take((size_t)D * D * 2);
  bf16_t* w1b   = (bf16_t*)take((size_t)H * FFD * hd * 2);
  bf16_t* w2b   = (bf16_t*)take((size_t)H * hd * FFD * 2);
  bf16_t* qb    = (bf16_t*)take((size_t)MT * D * 2);
  bf16_t* kb    = (bf16_t*)take((size_t)MT * D * 2);
  bf16_t* vTb   = (bf16_t*)take((size_t)MT * D * 2);  // [Bn,H,hd,S]
  bf16_t* ctxb  = (bf16_t*)take((size_t)MT * D * 2);
  float*  residf= (float*)take((size_t)MT * D * 4);
  float*  x1f   = (float*)take((size_t)MT * D * 4);
  bf16_t* x1b   = (bf16_t*)take((size_t)MT * D * 2);
  float*  fff   = (float*)take((size_t)MT * D * 4);

  const long nMD = MT * D;
  cvt_f32_bf16<<<dim3((unsigned)((nMD + 255) / 256)), 256, 0, stream>>>(x, xb, nMD);
  transpose_f32_bf16<<<dim3((D * D + 255) / 256), 256, 0, stream>>>(Wq, wqT, D, D);
  transpose_f32_bf16<<<dim3((D * D + 255) / 256), 256, 0, stream>>>(Wk, wkT, D, D);
  transpose_f32_bf16<<<dim3((D * D + 255) / 256), 256, 0, stream>>>(Wv, wvT, D, D);
  transpose_f32_bf16<<<dim3((D * D + 255) / 256), 256, 0, stream>>>(Wo, woT, D, D);
  cvt_f32_bf16<<<dim3((unsigned)((nMD + 255) / 256)), 256, 0, stream>>>(W1, w1b, (long)H * FFD * hd);
  cvt_f32_bf16<<<dim3((unsigned)((nMD + 255) / 256)), 256, 0, stream>>>(W2, w2b, (long)H * hd * FFD);

  // Q = x @ Wq + bq        (bf16 out [MT,D])
  gemm_wmma<2,4><<<dim3(D / 128, S / 64, Bn), 256, 0, stream>>>(
      xb, nullptr, D, (long)S * D, 0,
      wqT, D, 0, 0,
      nullptr, qb, D, (long)S * D, 0,
      bq, 1.0f, 0, 0, 0, D, 1);
  // K = x @ Wk + bk
  gemm_wmma<2,4><<<dim3(D / 128, S / 64, Bn), 256, 0, stream>>>(
      xb, nullptr, D, (long)S * D, 0,
      wkT, D, 0, 0,
      nullptr, kb, D, (long)S * D, 0,
      bk, 1.0f, 0, 0, 0, D, 1);
  // V = x @ Wv + bv, stored transposed -> [Bn,H,hd,S]
  gemm_wmma<2,4><<<dim3(D / 128, S / 64, Bn), 256, 0, stream>>>(
      xb, nullptr, D, (long)S * D, 0,
      wvT, D, 0, 0,
      nullptr, vTb, D, (long)D * S, 0,
      bv, 1.0f, 0, 1, (long)S, D, 1);
  // scores = (q @ k^T) / sqrt(hd) per (b,h) -> f32 into d_out attn region
  gemm_wmma<2,4><<<dim3(S / 128, S / 64, Bn * H), 256, 0, stream>>>(
      qb, nullptr, D, (long)S * D, hd,
      kb, D, (long)S * D, hd,
      attnOut, nullptr, S, (long)H * S * S, (long)S * S,
      nullptr, 0.125f, 0, 0, 0, hd, H);
  softmax_rows<<<dim3((unsigned)(Bn * H * S)), 256, 0, stream>>>(attnOut, S);
  // ctx = attn @ v per (b,h)  (A is f32 attn, converted on LDS fill) -> bf16 [MT,D]
  gemm_wmma<4,2><<<dim3(hd / 64, S / 128, Bn * H), 256, 0, stream>>>(
      nullptr, attnOut, S, (long)H * S * S, (long)S * S,
      vTb, S, (long)H * hd * S, (long)hd * S,
      nullptr, ctxb, D, (long)S * D, hd,
      nullptr, 1.0f, 0, 0, 0, S, H);
  // residual = ctx @ Wo + bo  (f32)
  gemm_wmma<2,4><<<dim3(D / 128, S / 64, Bn), 256, 0, stream>>>(
      ctxb, nullptr, D, (long)S * D, 0,
      woT, D, 0, 0,
      residf, nullptr, D, (long)S * D, 0,
      bo, 1.0f, 0, 0, 0, D, 1);
  // x1 = LN(x + residual)
  layernorm_rows<<<dim3((unsigned)MT), 256, 0, stream>>>(x, residf, g1, be1, x1f, x1b, D);
  // per-head FFN, fused (hidden stays in LDS)
  ff_fused<<<dim3((unsigned)(MT / 16), H), 256, 0, stream>>>(x1b, w1b, b1, w2b, b2, fff);
  // y = LN(x1 + ff)
  layernorm_rows<<<dim3((unsigned)MT), 256, 0, stream>>>(x1f, fff, g2, be2, yOut, nullptr, D);
}
